// RNNAttentionDecoder_55181739819608
// MI455X (gfx1250) — matrix operand order repaired and checked by reading, not compiled, not run on hardware
//
#include <hip/hip_runtime.h>

typedef __attribute__((ext_vector_type(16))) __bf16 v16bf;
typedef __attribute__((ext_vector_type(8)))  float  v8f;
typedef unsigned long long ull;

#define B_  32
#define D_  512
#define K_  256
#define DK_ 768
#define G_  2048
#define V_  10000
#define S_  196
#define T_  300

union FragB { v16bf v; uint4 q[2]; unsigned u[8]; };

static __device__ inline v8f wmma_bf16(v16bf a, v16bf b, v8f c) {
  return __builtin_amdgcn_wmma_f32_16x16x32_bf16(false, a, false, b, (short)0, c, false, false);
}

// B fragment: B[k][n] = W[n][k]; lane holds column n = nbase+(lane&15),
// k-run = kbase + (lane>>4)*16 .. +15  (contiguous in W's row -> two b128 loads)
static __device__ inline v16bf load_b_frag(const __bf16* W, int nbase, int Kd, int kbase, int lane) {
  FragB f;
  const uint4* p = (const uint4*)(W + (size_t)(nbase + (lane & 15)) * Kd + kbase + ((lane >> 4) << 4));
  f.q[0] = p[0];
  f.q[1] = p[1];
  return f.v;
}

// A fragment from LDS (row-major bf16 rows, W2 = width/2 uints per row).
// lane group kh=lane>>4: elems 0..15 -> K = kh*16 + e
static __device__ inline v16bf load_a_frag(const unsigned* Alds, int W2, int m, int lk, int lane) {
  FragB f;
  int base = m * W2 + (lk >> 1) + ((lane >> 4) << 3);
#pragma unroll
  for (int v = 0; v < 4; ++v) f.u[v] = Alds[base + v];
#pragma unroll
  for (int v = 4; v < 8; ++v) f.u[v] = Alds[base + v];
  return f.v;
}

static __device__ inline float sigm(float x) { return 1.0f / (1.0f + __expf(-x)); }

// ---------------- one-time weight conversion / fusion ----------------
__global__ __launch_bounds__(256) void k_convert(
    const float* x, const float* emb, const float* Wk, const float* Wv, const float* Wp,
    const float* Wih0, const float* Whh0, const float* bih0, const float* bhh0,
    const float* Wih1, const float* Whh1, const float* bih1, const float* bhh1,
    __bf16* x_bf, __bf16* emb_bf, __bf16* Wk_bf, __bf16* Wv_bf, __bf16* Wp_bf,
    __bf16* Wc0, __bf16* Wc1, float* b0, float* b1)
{
  const long S0 = 3211264, S1 = 5120000, S2 = 131072, S3 = 131072, S4 = 7680000,
             S5 = 2621440, S6 = 2097152, S7 = 2048, S8 = 2048;
  const long TOT = S0 + S1 + S2 + S3 + S4 + S5 + S6 + S7 + S8;
  long stride = (long)gridDim.x * 256;
  for (long i = (long)blockIdx.x * 256 + threadIdx.x; i < TOT; i += stride) {
    long r = i;
    if (r < S0) { x_bf[r]   = (__bf16)x[r];   continue; } r -= S0;
    if (r < S1) { emb_bf[r] = (__bf16)emb[r]; continue; } r -= S1;
    if (r < S2) { Wk_bf[r]  = (__bf16)Wk[r];  continue; } r -= S2;
    if (r < S3) { Wv_bf[r]  = (__bf16)Wv[r];  continue; } r -= S3;
    if (r < S4) { Wp_bf[r]  = (__bf16)Wp[r];  continue; } r -= S4;
    if (r < S5) { long n = r / 1280, k = r % 1280;
                  Wc0[r] = (__bf16)((k < 768) ? Wih0[n * 768 + k] : Whh0[n * 512 + (k - 768)]);
                  continue; } r -= S5;
    if (r < S6) { long n = r >> 10, k = r & 1023;
                  Wc1[r] = (__bf16)((k < 512) ? Wih1[n * 512 + k] : Whh1[n * 512 + (k - 512)]);
                  continue; } r -= S6;
    if (r < S7) { b0[r] = bih0[r] + bhh0[r]; continue; } r -= S7;
    b1[r] = bih1[r] + bhh1[r];
  }
}

// ---------------- state init ----------------
__global__ __launch_bounds__(256) void k_init(
    __bf16* h0a, __bf16* h0b, __bf16* h1a, __bf16* h1b,
    float* c0, float* c1, __bf16* ctx_bf, ull* packed)
{
  int i = blockIdx.x * 256 + threadIdx.x;
  if (i < 16384) {
    h0a[i] = (__bf16)0.0f; h0b[i] = (__bf16)0.0f;
    h1a[i] = (__bf16)0.0f; h1b[i] = (__bf16)0.0f;
    c0[i] = 0.0f; c1[i] = 0.0f;
  }
  if (i < 8192) ctx_bf[i] = (__bf16)0.0f;
  if (i < 32) { packed[i] = 0ull; packed[32 + i] = 0xFFFFFFFFull; } // tok0 = 0
}

// ---------------- keys/values precompute: (B*S,512) @ W.T -> (B*S,256) ----------------
__global__ __launch_bounds__(256) void k_kv(
    const __bf16* x_bf, const __bf16* Wk_bf, const __bf16* Wv_bf,
    const float* bk, const float* bv, float* keys, float* values)
{
  __shared__ unsigned Alds[32 * 256];
  const __bf16* W = blockIdx.y ? Wv_bf : Wk_bf;
  const float* bias = blockIdx.y ? bv : bk;
  float* out = blockIdx.y ? values : keys;
  int tid = threadIdx.x, lane = tid & 31, wave = tid >> 5;
  int mblock = blockIdx.x * 32;
  for (int idx = tid; idx < 32 * 256; idx += 256) {
    int m = idx >> 8, ku = idx & 255;
    Alds[idx] = *(const unsigned*)(x_bf + ((size_t)(mblock + m)) * 512 + (ku << 1));
  }
  __syncthreads();
#pragma unroll
  for (int i = 0; i < 4; ++i) {
    int id = wave * 4 + i, mtile = id >> 4, nt = id & 15;
    int nb = nt * 16;
    v8f acc = {};
    for (int lk = 0; lk < 512; lk += 32) {
      v16bf bf = load_b_frag(W, nb, 512, lk, lane);
      v16bf af = load_a_frag(Alds, 256, (lane & 15) + mtile * 16, lk, lane);
      acc = wmma_bf16(af, bf, acc);
    }
    int c = lane & 15, rb = (lane >> 4) << 3;
    float bs = bias[nb + c];
#pragma unroll
    for (int r = 0; r < 8; ++r) {
      int row = mblock + mtile * 16 + rb + r;
      out[(size_t)row * 256 + nb + c] = acc[r] + bs;
    }
  }
}

// ---------------- LSTM cell 0: A=[emb[tok]|ctx|h0_old] (32x1280), Wc0 (2048x1280) ----------------
__global__ __launch_bounds__(256) void k_lstm0(
    const __bf16* emb_bf, const __bf16* ctx_bf, const __bf16* h0_rd, __bf16* h0_wr,
    float* c0, const __bf16* Wc0, const float* b0, const ull* packed_prev, ull* packed_cur)
{
  __shared__ unsigned Alds[32 * 256];
  __shared__ float gbuf[32][128];
  __shared__ int toks[32];
  int tid = threadIdx.x, lane = tid & 31, wave = tid >> 5;
  if (tid < 32) {
    ull pv = packed_prev[tid];
    toks[tid] = (int)(0xFFFFFFFFu - (unsigned)(pv & 0xFFFFFFFFull));
    if (blockIdx.x == 0) packed_cur[tid] = 0ull;
  }
  int db = blockIdx.x;                 // d-block of 32 cols
  int j = wave, g = j >> 1;
  int nbase = g * 512 + db * 32 + (j & 1) * 16;
  v8f acc0 = {}, acc1 = {};
  const int so[4] = {0, 512, 1024, 1280};
  for (int s = 0; s < 3; ++s) {
    int k0 = so[s], width = so[s + 1] - k0, W2 = width >> 1;
    __syncthreads();
    for (int idx = tid; idx < 32 * W2; idx += 256) {
      int m = idx / W2, ku = idx % W2, k = k0 + (ku << 1);
      unsigned val;
      if (k < 512)      val = *(const unsigned*)(emb_bf + (size_t)toks[m] * 512 + k);
      else if (k < 768) val = *(const unsigned*)(ctx_bf + m * 256 + (k - 512));
      else              val = *(const unsigned*)(h0_rd + m * 512 + (k - 768));
      Alds[m * W2 + ku] = val;
    }
    __syncthreads();
    for (int lk = 0; lk < width; lk += 32) {
      v16bf bf = load_b_frag(Wc0, nbase, 1280, k0 + lk, lane);
      v16bf a0 = load_a_frag(Alds, W2, (lane & 15), lk, lane);
      v16bf a1 = load_a_frag(Alds, W2, (lane & 15) + 16, lk, lane);
      acc0 = wmma_bf16(a0, bf, acc0);
      acc1 = wmma_bf16(a1, bf, acc1);
    }
  }
  int col = j * 16 + (lane & 15), rb = (lane >> 4) << 3;
#pragma unroll
  for (int r = 0; r < 8; ++r) { gbuf[rb + r][col] = acc0[r]; gbuf[16 + rb + r][col] = acc1[r]; }
  __syncthreads();
  for (int idx = tid; idx < 1024; idx += 256) {
    int m = idx >> 5, dd = idx & 31, d = db * 32 + dd;
    float gi = gbuf[m][dd]      + b0[d];
    float gf = gbuf[m][32 + dd] + b0[512 + d];
    float gg = gbuf[m][64 + dd] + b0[1024 + d];
    float go = gbuf[m][96 + dd] + b0[1536 + d];
    float c = sigm(gf) * c0[m * 512 + d] + sigm(gi) * tanhf(gg);
    float h = sigm(go) * tanhf(c);
    c0[m * 512 + d] = c;
    h0_wr[m * 512 + d] = (__bf16)h;
  }
}

// ---------------- LSTM cell 1: A=[h0_new|h1_old] (32x1024), Wc1 (2048x1024) ----------------
__global__ __launch_bounds__(256) void k_lstm1(
    const __bf16* h0_new, const __bf16* h1_rd, __bf16* h1_wr,
    float* c1, const __bf16* Wc1, const float* b1)
{
  __shared__ unsigned Alds[32 * 256];
  __shared__ float gbuf[32][128];
  int tid = threadIdx.x, lane = tid & 31, wave = tid >> 5;
  int db = blockIdx.x;
  int j = wave, g = j >> 1;
  int nbase = g * 512 + db * 32 + (j & 1) * 16;
  v8f acc0 = {}, acc1 = {};
  for (int s = 0; s < 2; ++s) {
    int k0 = s * 512;
    __syncthreads();
    const __bf16* src = s ? h1_rd : h0_new;
    for (int idx = tid; idx < 32 * 256; idx += 256) {
      int m = idx >> 8, ku = idx & 255;
      Alds[idx] = *(const unsigned*)(src + m * 512 + (ku << 1));
    }
    __syncthreads();
    for (int lk = 0; lk < 512; lk += 32) {
      v16bf bf = load_b_frag(Wc1, nbase, 1024, k0 + lk, lane);
      v16bf a0 = load_a_frag(Alds, 256, (lane & 15), lk, lane);
      v16bf a1 = load_a_frag(Alds, 256, (lane & 15) + 16, lk, lane);
      acc0 = wmma_bf16(a0, bf, acc0);
      acc1 = wmma_bf16(a1, bf, acc1);
    }
  }
  int col = j * 16 + (lane & 15), rb = (lane >> 4) << 3;
#pragma unroll
  for (int r = 0; r < 8; ++r) { gbuf[rb + r][col] = acc0[r]; gbuf[16 + rb + r][col] = acc1[r]; }
  __syncthreads();
  for (int idx = tid; idx < 1024; idx += 256) {
    int m = idx >> 5, dd = idx & 31, d = db * 32 + dd;
    float gi = gbuf[m][dd]      + b1[d];
    float gf = gbuf[m][32 + dd] + b1[512 + d];
    float gg = gbuf[m][64 + dd] + b1[1024 + d];
    float go = gbuf[m][96 + dd] + b1[1536 + d];
    float c = sigm(gf) * c1[m * 512 + d] + sigm(gi) * tanhf(gg);
    float h = sigm(go) * tanhf(c);
    c1[m * 512 + d] = c;
    h1_wr[m * 512 + d] = (__bf16)h;
  }
}

// ---------------- attention: q = h1 Wq.T + bq; softmax(keys.q); ctx = attn.values ----------------
__global__ __launch_bounds__(256) void k_attn(
    const __bf16* h1_new, const float* Wq, const float* bq,
    const float* keys, const float* values, __bf16* ctx_bf)
{
  int b = blockIdx.x, tid = threadIdx.x;
  __shared__ float h1f[512];
  __shared__ float qv[256];
  __shared__ float en[256];
  __shared__ float red[256];
  for (int k = tid; k < 512; k += 256) h1f[k] = (float)h1_new[b * 512 + k];
  __syncthreads();
  {
    float acc = bq[tid];
    const float* wr = Wq + (size_t)tid * 512;
    for (int k = 0; k < 512; ++k) acc += wr[k] * h1f[k];
    qv[tid] = acc;
  }
  __syncthreads();
  float e = -__builtin_inff();
  if (tid < 196) {
    const float* kr = keys + ((size_t)b * 196 + tid) * 256;
    float acc = 0.0f;
    for (int k = 0; k < 256; ++k) acc += kr[k] * qv[k];
    e = acc * 0.0625f;  // 1/sqrt(256)
  }
  en[tid] = e;
  red[tid] = e; __syncthreads();
  for (int s = 128; s > 0; s >>= 1) { if (tid < s) red[tid] = fmaxf(red[tid], red[tid + s]); __syncthreads(); }
  float mx = red[0]; __syncthreads();
  float ex = (tid < 196) ? __expf(en[tid] - mx) : 0.0f;
  en[tid] = ex;
  red[tid] = ex; __syncthreads();
  for (int s = 128; s > 0; s >>= 1) { if (tid < s) red[tid] += red[tid + s]; __syncthreads(); }
  float inv = 1.0f / red[0];
  __syncthreads();
  {
    float acc = 0.0f;
    const float* vb = values + (size_t)b * 196 * 256 + tid;
    for (int s = 0; s < 196; ++s) acc += en[s] * vb[s * 256];
    ctx_bf[b * 256 + tid] = (__bf16)(acc * inv);
  }
}

// ---------------- pred: A=[h1|ctx] (32x768) @ Wp.T (10000x768) + argmax ----------------
__global__ __launch_bounds__(256) void k_pred(
    const __bf16* h1_new, const __bf16* ctx_bf, const __bf16* Wp_bf, const float* bp,
    float* out, int t, ull* packed_cur)
{
  __shared__ unsigned Alds[32 * 256];
  __shared__ float pbuf[32][128];
  __shared__ ull cand[32][8];
  int tid = threadIdx.x, lane = tid & 31, wave = tid >> 5;
  int n0 = blockIdx.x * 128;
  int nb = n0 + wave * 16;
  bool valid = nb < V_;   // wave-uniform
  v8f acc0 = {}, acc1 = {};
  const int so[3] = {0, 512, 768};
  for (int s = 0; s < 2; ++s) {
    int k0 = so[s], width = so[s + 1] - k0, W2 = width >> 1;
    __syncthreads();
    for (int idx = tid; idx < 32 * W2; idx += 256) {
      int m = idx / W2, ku = idx % W2, k = k0 + (ku << 1);
      unsigned val = (k < 512) ? *(const unsigned*)(h1_new + m * 512 + k)
                               : *(const unsigned*)(ctx_bf + m * 256 + (k - 512));
      Alds[m * W2 + ku] = val;
    }
    __syncthreads();
    if (valid) {
      for (int lk = 0; lk < width; lk += 32) {
        __builtin_prefetch(Wp_bf + (size_t)(nb + (lane & 15)) * 768 + k0 + lk + 64, 0, 1);
        v16bf bf = load_b_frag(Wp_bf, nb, 768, k0 + lk, lane);
        v16bf a0 = load_a_frag(Alds, W2, (lane & 15), lk, lane);
        v16bf a1 = load_a_frag(Alds, W2, (lane & 15) + 16, lk, lane);
        acc0 = wmma_bf16(a0, bf, acc0);
        acc1 = wmma_bf16(a1, bf, acc1);
      }
    }
  }
  int c = lane & 15, rb = (lane >> 4) << 3;
  int col = wave * 16 + c;
  if (valid) {
    float bias = bp[nb + c];
#pragma unroll
    for (int r = 0; r < 8; ++r) {
      float v0 = acc0[r] + bias, v1 = acc1[r] + bias;
      pbuf[rb + r][col] = v0;
      pbuf[16 + rb + r][col] = v1;
      out[((size_t)(rb + r) * T_ + t) * V_ + nb + c] = v0;
      out[((size_t)(16 + rb + r) * T_ + t) * V_ + nb + c] = v1;
    }
  } else {
#pragma unroll
    for (int r = 0; r < 8; ++r) { pbuf[rb + r][col] = -__builtin_inff(); pbuf[16 + rb + r][col] = -__builtin_inff(); }
  }
  __syncthreads();
  {
    int m = tid >> 3, seg = tid & 7;
    float best = -__builtin_inff(); int bi = 0;
    for (int cc = 0; cc < 16; ++cc) {
      float v = pbuf[m][seg * 16 + cc];
      if (v > best) { best = v; bi = n0 + seg * 16 + cc; }
    }
    unsigned ub = __float_as_uint(best);
    ub = (ub & 0x80000000u) ? ~ub : (ub | 0x80000000u);     // order-preserving map
    cand[m][seg] = (((ull)ub) << 32) | (ull)(0xFFFFFFFFu - (unsigned)bi); // ties -> lowest index
  }
  __syncthreads();
  if (tid < 32) {
    ull best = cand[tid][0];
#pragma unroll
    for (int s = 1; s < 8; ++s) { ull v = cand[tid][s]; best = (v > best) ? v : best; }
    atomicMax(&packed_cur[tid], best);
  }
}

// ---------------- host ----------------
extern "C" void kernel_launch(void* const* d_in, const int* in_sizes, int n_in,
                              void* d_out, int out_size, void* d_ws, size_t ws_size,
                              hipStream_t stream) {
  const float* x    = (const float*)d_in[0];
  const float* emb  = (const float*)d_in[1];
  const float* Wk   = (const float*)d_in[2];
  const float* bk   = (const float*)d_in[3];
  const float* Wv   = (const float*)d_in[4];
  const float* bv   = (const float*)d_in[5];
  const float* Wq   = (const float*)d_in[6];
  const float* bq   = (const float*)d_in[7];
  const float* Wp   = (const float*)d_in[8];
  const float* bp   = (const float*)d_in[9];
  const float* Wih0 = (const float*)d_in[10];
  const float* Whh0 = (const float*)d_in[11];
  const float* bih0 = (const float*)d_in[12];
  const float* bhh0 = (const float*)d_in[13];
  const float* Wih1 = (const float*)d_in[14];
  const float* Whh1 = (const float*)d_in[15];
  const float* bih1 = (const float*)d_in[16];
  const float* bhh1 = (const float*)d_in[17];
  float* out = (float*)d_out;

  size_t off = 0;
  char* base = (char*)d_ws;
  auto carve = [&](size_t bytes) -> void* {
    void* r = base + off;
    off = (off + bytes + 255) & ~(size_t)255;
    return r;
  };
  __bf16* x_bf   = (__bf16*)carve(3211264 * 2);
  __bf16* emb_bf = (__bf16*)carve(5120000 * 2);
  __bf16* Wk_bf  = (__bf16*)carve(131072 * 2);
  __bf16* Wv_bf  = (__bf16*)carve(131072 * 2);
  __bf16* Wp_bf  = (__bf16*)carve(7680000 * 2);
  __bf16* Wc0    = (__bf16*)carve(2621440 * 2);
  __bf16* Wc1    = (__bf16*)carve(2097152 * 2);
  float*  b0     = (float*)carve(2048 * 4);
  float*  b1     = (float*)carve(2048 * 4);
  float*  keys   = (float*)carve((size_t)6272 * 256 * 4);
  float*  values = (float*)carve((size_t)6272 * 256 * 4);
  __bf16* h0a    = (__bf16*)carve(16384 * 2);
  __bf16* h0b    = (__bf16*)carve(16384 * 2);
  __bf16* h1a    = (__bf16*)carve(16384 * 2);
  __bf16* h1b    = (__bf16*)carve(16384 * 2);
  float*  c0     = (float*)carve(16384 * 4);
  float*  c1     = (float*)carve(16384 * 4);
  __bf16* ctxb   = (__bf16*)carve(8192 * 2);
  ull*    packed = (ull*)carve(64 * 8);

  k_convert<<<4096, 256, 0, stream>>>(x, emb, Wk, Wv, Wp, Wih0, Whh0, bih0, bhh0,
                                      Wih1, Whh1, bih1, bhh1,
                                      x_bf, emb_bf, Wk_bf, Wv_bf, Wp_bf, Wc0, Wc1, b0, b1);
  k_init<<<64, 256, 0, stream>>>(h0a, h0b, h1a, h1b, c0, c1, ctxb, packed);
  k_kv<<<dim3(196, 2), 256, 0, stream>>>(x_bf, Wk_bf, Wv_bf, bk, bv, keys, values);

  for (int t = 0; t < T_; ++t) {
    const ull* prev = packed + ((t + 1) & 1) * 32;
    ull* cur = packed + (t & 1) * 32;
    const __bf16* h0_rd = (t & 1) ? h0b : h0a;
    __bf16* h0_wr = (t & 1) ? h0a : h0b;
    const __bf16* h1_rd = (t & 1) ? h1b : h1a;
    __bf16* h1_wr = (t & 1) ? h1a : h1b;
    k_lstm0<<<16, 256, 0, stream>>>(emb_bf, ctxb, h0_rd, h0_wr, c0, Wc0, b0, prev, cur);
    k_lstm1<<<16, 256, 0, stream>>>(h0_wr, h1_rd, h1_wr, c1, Wc1, b1);
    k_attn<<<32, 256, 0, stream>>>(h1_wr, Wq, bq, keys, values, ctxb);
    k_pred<<<79, 256, 0, stream>>>(h1_wr, ctxb, Wp_bf, bp, out, t, cur);
  }
}
